// GPT2D_71949292143281
// MI455X (gfx1250) — matrix operand order repaired and checked by reading, not compiled, not run on hardware
//
#include <hip/hip_runtime.h>

#define VOCAB  50257
#define NEMBD  768
#define SEQ    1024
#define NROWS  8192              // BATCH * SEQ
#define MROWS  64                // rows per block
#define XPAD   8
#define XSTRIDE (NEMBD + XPAD)   // 776 -> staggers LDS banks across rows
#define KSTEPS (NEMBD / 32)      // 24
#define NTILES ((VOCAB + 127) / 128)           // 393 iterations of N=128
#define VSPLIT 4
#define ITERS_PER_SPLIT ((NTILES + VSPLIT - 1) / VSPLIT)  // 99

typedef __bf16  bf16_t;
typedef bf16_t  bf16x8  __attribute__((ext_vector_type(8)));
typedef bf16_t  bf16x16 __attribute__((ext_vector_type(16)));
typedef float   f32x8   __attribute__((ext_vector_type(8)));

union AFrag { bf16x16 v; bf16x8 h[2]; };

__device__ __forceinline__ unsigned short f2bf(float f) {
    unsigned u = __float_as_uint(f);
    u += 0x7FFFu + ((u >> 16) & 1u);           // round-to-nearest-even
    return (unsigned short)(u >> 16);
}

__device__ __forceinline__ float bf_round(float f) {   // f32 -> bf16 -> f32
    return __uint_as_float((unsigned)f2bf(f) << 16);
}

__device__ __forceinline__ float gelu_tanh(float x) {
    float x3 = x * x * x;
    float t  = tanhf(0.7978845608028654f * (x + 0.044715f * x3));
    return 0.5f * x * (1.0f + t);
}

// ---------- W f32 -> bf16 pre-conversion (one pass over 154 MB) ----------
__global__ __launch_bounds__(256) void wconv_kernel(const float* __restrict__ W,
                                                    unsigned short* __restrict__ O,
                                                    int n4) {
    int i      = blockIdx.x * blockDim.x + threadIdx.x;
    int stride = gridDim.x * blockDim.x;
    const float4* W4 = (const float4*)W;
    ushort4*      O4 = (ushort4*)O;
    for (; i < n4; i += stride) {
        float4 f = W4[i];
        ushort4 o;
        o.x = f2bf(f.x); o.y = f2bf(f.y); o.z = f2bf(f.z); o.w = f2bf(f.w);
        O4[i] = o;
    }
}

// B-fragment loader: 32x16 bf16 tile, lane holds N=nsub, K = khalf..khalf+15 (32B contiguous)
template <bool WBF>
__device__ __forceinline__ bf16x16 load_b(const unsigned short* __restrict__ bp,
                                          const float* __restrict__ fp, int kofs) {
    if constexpr (WBF) {
        return *(const bf16x16*)(bp + kofs);
    } else {
        const float* wp = fp + kofs;
        union { bf16x16 v; unsigned short e[16]; } bu;
        #pragma unroll
        for (int i = 0; i < 16; ++i) bu.e[i] = f2bf(wp[i]);
        return bu.v;
    }
}

__device__ __forceinline__ void load_a(AFrag* a,
                                       const unsigned short* __restrict__ apLo,
                                       const unsigned short* __restrict__ apHi,
                                       int k0) {
    a[0].h[0] = *(const bf16x8*)(apLo + k0);
    a[0].h[1] = *(const bf16x8*)(apLo + k0 + 8);
    a[1].h[0] = *(const bf16x8*)(apLo + 16 * XSTRIDE + k0);
    a[1].h[1] = *(const bf16x8*)(apLo + 16 * XSTRIDE + k0 + 8);
    a[2].h[0] = *(const bf16x8*)(apHi + k0);
    a[2].h[1] = *(const bf16x8*)(apHi + k0 + 8);
    a[3].h[0] = *(const bf16x8*)(apHi + 16 * XSTRIDE + k0);
    a[3].h[1] = *(const bf16x8*)(apHi + 16 * XSTRIDE + k0 + 8);
}

// ---------- fused embed+GELU -> bf16 WMMA GEMM -> online softmax ----------
template <bool WBF>
__global__ __launch_bounds__(256)
void fused_gemm_softmax(const int*   __restrict__ data,
                        const float* __restrict__ wte,
                        const float* __restrict__ wpe,
                        const float* __restrict__ Wf,
                        const unsigned short* __restrict__ Wbf,
                        float* __restrict__ pmax,   // [VSPLIT][NROWS]
                        float* __restrict__ psum)   // [VSPLIT][NROWS]
{
    __shared__ __align__(16) unsigned short xs[MROWS][XSTRIDE]; // ~97 KB of 320 KB WGP LDS
    __shared__ float wmaxs[8][MROWS];
    __shared__ float wsums[8][MROWS];

    const int tid   = threadIdx.x;
    const int rb    = blockIdx.x >> 2;          // row block 0..127
    const int split = blockIdx.x & (VSPLIT - 1);
    const int m0    = rb * MROWS;

    // stage x = gelu(wte[tok] + wpe[pos]) into LDS as bf16
    for (int rr = 0; rr < MROWS; ++rr) {
        int m   = m0 + rr;
        int tok = data[m];
        int pos = m & (SEQ - 1);
        const float* wt = wte + (size_t)tok * NEMBD;
        const float* wp = wpe + (size_t)pos * NEMBD;
        for (int j = tid; j < NEMBD; j += 256)
            xs[rr][j] = f2bf(gelu_tanh(wt[j] + wp[j]));
    }
    __syncthreads();

    const int lane = tid & 31;
    const int wv   = tid >> 5;        // wave id 0..7 -> owns N-subtile
    const int nsub = lane & 15;       // N (for B/C) and M (for A) within tile
    const int half = lane >> 4;       // K-half selector per ISA layouts
    const int khalf = half * 16;

    // Two A base pointers keep all DS offsets < 64KB (16-bit DS offset field)
    const unsigned short* apLo = &xs[nsub][half * 16];       // mt 0,1
    const unsigned short* apHi = &xs[32 + nsub][half * 16];  // mt 2,3

    // Per-LANE running softmax state (flash style): each lane tracks only its
    // own logit columns -> no cross-lane ops inside the vocab loop.
    float rmax[4][8], rsum[4][8];
    #pragma unroll
    for (int mt = 0; mt < 4; ++mt)
        #pragma unroll
        for (int r = 0; r < 8; ++r) { rmax[mt][r] = -1e30f; rsum[mt][r] = 0.0f; }

    const int it0 = split * ITERS_PER_SPLIT;
    const int it1 = (it0 + ITERS_PER_SPLIT < NTILES) ? (it0 + ITERS_PER_SPLIT) : NTILES;

    for (int it = it0; it < it1; ++it) {
        // Compiler memory barrier: the xs tile is loop-invariant here, and
        // without this LLVM hoists ALL k-step ds_loads out of the vocab loop
        // (seen twice: 768 VGPRs of A -> scratch spills). Zero instructions.
        asm volatile("" ::: "memory");

        const int  v0    = it * 128;
        const int  ncol  = v0 + wv * 16 + nsub;           // this lane's logit column
        const bool valid = ncol < VOCAB;
        const int  nn    = valid ? ncol : (VOCAB - 1);    // clamp for safe addressing

        const unsigned short* bp = WBF ? (Wbf + (size_t)nn * NEMBD + khalf) : nullptr;
        const float*          fp = WBF ? nullptr : (Wf + (size_t)nn * NEMBD + khalf);

        f32x8 acc[4];
        #pragma unroll
        for (int mt = 0; mt < 4; ++mt)
            #pragma unroll
            for (int r = 0; r < 8; ++r) acc[mt][r] = 0.0f;

        // Explicit modulo double-buffering of A (LDS) and B (global) register
        // stages; k-loop kept rolled (cap 2) so stage indices fold to constants.
        AFrag   a[2][4];
        bf16x16 b[2];
        load_a(a[0], apLo, apHi, 0);
        b[0] = load_b<WBF>(bp, fp, 0);

        #pragma unroll 2
        for (int i = 0; i < KSTEPS - 1; ++i) {
            const int cur = i & 1;
            const int nxt = cur ^ 1;
            load_a(a[nxt], apLo, apHi, (i + 1) * 32);      // 8 ds_load_b128, one clause
            b[nxt] = load_b<WBF>(bp, fp, (i + 1) * 32);    // global loads fly during WMMAs
            acc[0] = __builtin_amdgcn_wmma_f32_16x16x32_bf16(false, a[cur][0].v, false, b[cur], (short)0, acc[0], false, false);
            acc[1] = __builtin_amdgcn_wmma_f32_16x16x32_bf16(false, a[cur][1].v, false, b[cur], (short)0, acc[1], false, false);
            acc[2] = __builtin_amdgcn_wmma_f32_16x16x32_bf16(false, a[cur][2].v, false, b[cur], (short)0, acc[2], false, false);
            acc[3] = __builtin_amdgcn_wmma_f32_16x16x32_bf16(false, a[cur][3].v, false, b[cur], (short)0, acc[3], false, false);
        }
        {   // peeled final k-step: stage (KSTEPS-1)&1 == 1
            constexpr int cur = (KSTEPS - 1) & 1;
            acc[0] = __builtin_amdgcn_wmma_f32_16x16x32_bf16(false, a[cur][0].v, false, b[cur], (short)0, acc[0], false, false);
            acc[1] = __builtin_amdgcn_wmma_f32_16x16x32_bf16(false, a[cur][1].v, false, b[cur], (short)0, acc[1], false, false);
            acc[2] = __builtin_amdgcn_wmma_f32_16x16x32_bf16(false, a[cur][2].v, false, b[cur], (short)0, acc[2], false, false);
            acc[3] = __builtin_amdgcn_wmma_f32_16x16x32_bf16(false, a[cur][3].v, false, b[cur], (short)0, acc[3], false, false);
        }

        // per-lane online softmax update: no shuffles, no LDS, no branches
        #pragma unroll
        for (int mt = 0; mt < 4; ++mt) {
            #pragma unroll
            for (int r = 0; r < 8; ++r) {
                float c  = valid ? acc[mt][r] : -1e30f;
                float m  = rmax[mt][r];
                float nm = fmaxf(m, c);
                rsum[mt][r] = rsum[mt][r] * __expf(m - nm) + __expf(c - nm);
                rmax[mt][r] = nm;
            }
        }
    }

    // merge (m,s) across the 16 lanes of each half ONCE, then across waves
    #pragma unroll
    for (int mt = 0; mt < 4; ++mt) {
        #pragma unroll
        for (int r = 0; r < 8; ++r) {
            float m = rmax[mt][r];
            float s = rsum[mt][r];
            #pragma unroll
            for (int d = 1; d <= 8; d <<= 1) {
                float mo = __shfl_xor(m, d, 32);
                float so = __shfl_xor(s, d, 32);
                float nm = fmaxf(m, mo);
                s = s * __expf(m - nm) + so * __expf(mo - nm);
                m = nm;
            }
            if (nsub == 0) {
                int rowLocal = mt * 16 + r + half * 8;
                wmaxs[wv][rowLocal] = m;
                wsums[wv][rowLocal] = s;
            }
        }
    }
    __syncthreads();
    if (tid < MROWS) {
        float M = -1e30f;
        #pragma unroll
        for (int w = 0; w < 8; ++w) M = fmaxf(M, wmaxs[w][tid]);
        float S = 0.0f;
        #pragma unroll
        for (int w = 0; w < 8; ++w) S += wsums[w][tid] * __expf(wmaxs[w][tid] - M);
        pmax[(size_t)split * NROWS + m0 + tid] = M;
        psum[(size_t)split * NROWS + m0 + tid] = S;
    }
}

// ---------- combine: label logit (direct dot), V-split merge, mean NLL ----------
// One wave per row: recompute x[row] = gelu(wte+wpe), dot with W[label[row]]
// (both operands rounded through bf16 to match WMMA operand precision).
__global__ __launch_bounds__(256)
void combine_loss(const int*   __restrict__ data,
                  const int*   __restrict__ label,
                  const float* __restrict__ wte,
                  const float* __restrict__ wpe,
                  const float* __restrict__ Wf,
                  const float* __restrict__ pmax,
                  const float* __restrict__ psum,
                  float* __restrict__ out) {
    const int lane = threadIdx.x & 31;
    const int row  = blockIdx.x * (blockDim.x >> 5) + (threadIdx.x >> 5);
    if (row >= NROWS) return;

    const int tok = data[row];
    const int pos = row & (SEQ - 1);
    const int lab = label[row];
    const float* wt = wte + (size_t)tok * NEMBD;
    const float* wp = wpe + (size_t)pos * NEMBD;
    const float* wl = Wf  + (size_t)lab * NEMBD;

    float dot = 0.0f;
    for (int j = lane; j < NEMBD; j += 32) {
        float xv = bf_round(gelu_tanh(wt[j] + wp[j]));
        float wv = bf_round(wl[j]);
        dot += xv * wv;
    }
    #pragma unroll
    for (int d = 1; d <= 16; d <<= 1) dot += __shfl_xor(dot, d, 32);

    if (lane == 0) {
        float M = -1e30f;
        #pragma unroll
        for (int s = 0; s < VSPLIT; ++s) M = fmaxf(M, pmax[s * NROWS + row]);
        float S = 0.0f;
        #pragma unroll
        for (int s = 0; s < VSPLIT; ++s) S += psum[s * NROWS + row] * __expf(pmax[s * NROWS + row] - M);
        float loss = M + __logf(S) - dot;
        atomicAdd(out, loss * (1.0f / (float)NROWS));
    }
}

extern "C" void kernel_launch(void* const* d_in, const int* in_sizes, int n_in,
                              void* d_out, int out_size, void* d_ws, size_t ws_size,
                              hipStream_t stream) {
    const int*   data  = (const int*)d_in[0];
    const int*   label = (const int*)d_in[1];
    const float* wte   = (const float*)d_in[2];
    const float* wpe   = (const float*)d_in[3];
    const float* Wf    = (const float*)d_in[4];
    float*       out   = (float*)d_out;

    // workspace layout: pmax[4][8192] | psum[4][8192] | Wbf16[VOCAB*NEMBD]
    float*          pmax = (float*)d_ws;
    float*          psum = pmax + (size_t)VSPLIT * NROWS;
    unsigned short* wbf  = (unsigned short*)(psum + (size_t)VSPLIT * NROWS); // 32B-aligned offset
    const size_t partial_bytes = (size_t)NROWS * (2 * VSPLIT) * sizeof(float);
    const size_t need_wbf      = partial_bytes + (size_t)VOCAB * NEMBD * sizeof(unsigned short);

    hipMemsetAsync(out, 0, (size_t)out_size * sizeof(float), stream);

    if (ws_size >= need_wbf) {
        int n4 = VOCAB * NEMBD / 4;
        wconv_kernel<<<dim3(2048), dim3(256), 0, stream>>>(Wf, wbf, n4);
        fused_gemm_softmax<true><<<dim3(128 * VSPLIT), dim3(256), 0, stream>>>(
            data, wte, wpe, Wf, wbf, pmax, psum);
    } else {
        fused_gemm_softmax<false><<<dim3(128 * VSPLIT), dim3(256), 0, stream>>>(
            data, wte, wpe, Wf, nullptr, pmax, psum);
    }
    // one wave per row: 8 rows per 256-thread block
    combine_loss<<<dim3(NROWS / 8), dim3(256), 0, stream>>>(
        data, label, wte, wpe, Wf, pmax, psum, out);
}
